// MLA_32830730011151
// MI455X (gfx1250) — matrix operand order repaired and checked by reading, not compiled, not run on hardware
//
#include <hip/hip_runtime.h>
#include <stdint.h>

// ---------------------------------------------------------------------------
// Types for CDNA5 WMMA (wave32, v_wmma_f32_16x16x32_bf16)
// ---------------------------------------------------------------------------
typedef __attribute__((ext_vector_type(16))) __bf16 v16bf;
typedef __attribute__((ext_vector_type(8)))  float  v8f;

struct __attribute__((aligned(16))) U4  { unsigned int x, y, z, w; };
struct __attribute__((aligned(16))) F4  { float x, y, z, w; };
struct __attribute__((aligned(8)))  US4 { unsigned short x, y, z, w; };
struct __attribute__((aligned(16))) US8 { unsigned short s[8]; };

union AFrag { v16bf v; U4 q[2]; };

__device__ __forceinline__ unsigned short f2b(float f) {
  unsigned int u = __float_as_uint(f);
  u += 0x7FFFu + ((u >> 16) & 1u);          // round-to-nearest-even bf16
  return (unsigned short)(u >> 16);
}

__device__ __forceinline__ v8f wmma_bf16(const AFrag& a, const AFrag& b, v8f c) {
  return __builtin_amdgcn_wmma_f32_16x16x32_bf16(
      false, a.v, false, b.v, (short)0, c, false, false);
}

// Async global->LDS 16B DMA (gfx1250, ASYNCcnt-tracked), per active lane.
__device__ __forceinline__ void async_ld_b128(unsigned lds_off, const void* g) {
  unsigned long long ga = (unsigned long long)(uintptr_t)g;
  asm volatile("global_load_async_to_lds_b128 %0, %1, off"
               :: "v"(lds_off), "v"(ga) : "memory");
}
__device__ __forceinline__ void wait_async0() {
  asm volatile("s_wait_asynccnt 0x0" ::: "memory");
}

// ---------------------------------------------------------------------------
// Generic tiled GEMM: C[M,N] = A[M,K] * B[K,N]
//   A: fp32 (A_BF16=0) or bf16 (A_BF16=1), row-major, K % 64 == 0, M % 128 == 0
//   B: fp32 row-major (converted to bf16 in LDS, stored transposed [n][k])
//   MODE 0: C fp32 row-major.  MODE 1: bf16 scatter into head-major layout
//           dst[((b*H + h)*S + s)*dstride + d],  h = col/128, d = col%128
// Block: 256 thr = 8 waves in 4(M) x 2(N) grid; wave = 32x64 = 8 WMMA tiles;
// BK=64 -> 16 v_wmma per barrier pair.
// ---------------------------------------------------------------------------
template <int A_BF16, int MODE>
__global__ __launch_bounds__(256) void gemm_wmma_bf16(
    const void* __restrict__ Aptr,
    const float* __restrict__ Bmat,
    float* __restrict__ Cf, unsigned short* __restrict__ Cb,
    int M, int N, int K, int Ssz, int Hh, int dstride)
{
  __shared__ unsigned short As[128 * 64];   // [m][k]
  __shared__ unsigned short Bs[128 * 64];   // [n][k]  (transposed)
  const int tid  = threadIdx.x;
  const int lane = tid & 31, w = tid >> 5;
  const int wm = w & 3, wn = w >> 2;
  const int hfw = lane >> 4, lr = lane & 15;
  const int m0 = blockIdx.y * 128, n0 = blockIdx.x * 128;

  v8f acc[2][4] = {};

  for (int k0 = 0; k0 < K; k0 += 64) {
    if (k0 + 64 < K) {                      // prefetch next tiles into GL2
      __builtin_prefetch((const char*)Aptr +
          (((size_t)(m0 + (tid >> 1)) * K + k0 + 64 + (tid & 1) * 32) << (A_BF16 ? 1 : 2)), 0, 1);
      __builtin_prefetch(Bmat + (size_t)(k0 + 64 + (tid >> 2)) * N + n0 + (tid & 3) * 32, 0, 1);
    }
#pragma unroll
    for (int i = 0; i < 8; ++i) {                       // A tile 128x64
      int idx = tid * 8 + i;
      int row = idx >> 4, kc = (idx & 15) << 2;
      size_t ga = (size_t)(m0 + row) * K + k0 + kc;
      unsigned short* dst = &As[row * 64 + kc];
      if (A_BF16) {
        US4 v = *(const US4*)((const unsigned short*)Aptr + ga);
        dst[0] = v.x; dst[1] = v.y; dst[2] = v.z; dst[3] = v.w;
      } else {
        F4 v = *(const F4*)((const float*)Aptr + ga);
        dst[0] = f2b(v.x); dst[1] = f2b(v.y); dst[2] = f2b(v.z); dst[3] = f2b(v.w);
      }
    }
#pragma unroll
    for (int i = 0; i < 8; ++i) {                       // B tile 64x128 -> [n][k]
      int idx = tid * 8 + i;
      int kr = idx >> 5, nc = (idx & 31) << 2;
      F4 v = {0.f, 0.f, 0.f, 0.f};
      if (n0 + nc < N) v = *(const F4*)(Bmat + (size_t)(k0 + kr) * N + n0 + nc);
      Bs[(nc + 0) * 64 + kr] = f2b(v.x);
      Bs[(nc + 1) * 64 + kr] = f2b(v.y);
      Bs[(nc + 2) * 64 + kr] = f2b(v.z);
      Bs[(nc + 3) * 64 + kr] = f2b(v.w);
    }
    __syncthreads();

#pragma unroll
    for (int ks = 0; ks < 64; ks += 32) {
      AFrag a[2], b[4];
#pragma unroll
      for (int tm = 0; tm < 2; ++tm) {
        int row = wm * 32 + tm * 16 + lr;
        a[tm].q[0] = *(const U4*)(As + row * 64 + ks + hfw * 8);
        a[tm].q[1] = *(const U4*)(As + row * 64 + ks + 16 + hfw * 8);
      }
#pragma unroll
      for (int tn = 0; tn < 4; ++tn) {
        int col = wn * 64 + tn * 16 + lr;
        b[tn].q[0] = *(const U4*)(Bs + col * 64 + ks + hfw * 8);
        b[tn].q[1] = *(const U4*)(Bs + col * 64 + ks + 16 + hfw * 8);
      }
#pragma unroll
      for (int tm = 0; tm < 2; ++tm)
#pragma unroll
        for (int tn = 0; tn < 4; ++tn)
          acc[tm][tn] = wmma_bf16(a[tm], b[tn], acc[tm][tn]);
    }
    __syncthreads();
  }

#pragma unroll
  for (int tm = 0; tm < 2; ++tm)
#pragma unroll
    for (int tn = 0; tn < 4; ++tn)
#pragma unroll
      for (int r = 0; r < 8; ++r) {
        int row = m0 + wm * 32 + tm * 16 + hfw * 8 + r;
        int col = n0 + wn * 64 + tn * 16 + lr;
        if (col < N) {
          float val = acc[tm][tn][r];
          if (MODE == 0) {
            Cf[(size_t)row * N + col] = val;
          } else {
            int bb = row / Ssz, s = row % Ssz;
            int hh = col >> 7, d = col & 127;
            Cb[(((size_t)bb * Hh + hh) * Ssz + s) * dstride + d] = f2b(val);
          }
        }
      }
}

// ---------------------------------------------------------------------------
// RoPE: pairs (2i, 2i+1), theta_i = 10000^(-2i/64), write bf16 into head-major
// K/Q buffers at dim offset 128 (rope dims of the 192-wide head).
// ---------------------------------------------------------------------------
__global__ void rope_k_kernel(const float* __restrict__ KR,
                              unsigned short* __restrict__ Kf,
                              int S, int H, int total)
{
  int idx = blockIdx.x * blockDim.x + threadIdx.x;
  if (idx >= total) return;
  int i = idx & 31, tok = idx >> 5;
  int s = tok % S, b = tok / S;
  float theta = __powf(10000.0f, -((float)(2 * i)) / 64.0f);
  float sn, cs; __sincosf((float)s * theta, &sn, &cs);
  float x0 = KR[(size_t)tok * 64 + 2 * i];
  float x1 = KR[(size_t)tok * 64 + 2 * i + 1];
  unsigned short o0 = f2b(x0 * cs - x1 * sn);
  unsigned short o1 = f2b(x1 * cs + x0 * sn);
  for (int h = 0; h < H; ++h) {                 // rope-K broadcast to all heads
    size_t base = (((size_t)b * H + h) * S + s) * 192 + 128 + 2 * i;
    Kf[base] = o0; Kf[base + 1] = o1;
  }
}

__global__ void rope_q_kernel(const float* __restrict__ QR,
                              unsigned short* __restrict__ Qf,
                              int S, int H, int total)
{
  int idx = blockIdx.x * blockDim.x + threadIdx.x;
  if (idx >= total) return;
  int i = idx & 31, t2 = idx >> 5;
  int h = t2 % H, tok = t2 / H;
  int s = tok % S, b = tok / S;
  float theta = __powf(10000.0f, -((float)(2 * i)) / 64.0f);
  float sn, cs; __sincosf((float)s * theta, &sn, &cs);
  float x0 = QR[(size_t)tok * (H * 64) + h * 64 + 2 * i];
  float x1 = QR[(size_t)tok * (H * 64) + h * 64 + 2 * i + 1];
  size_t base = (((size_t)b * H + h) * S + s) * 192 + 128 + 2 * i;
  Qf[base]     = f2b(x0 * cs - x1 * sn);
  Qf[base + 1] = f2b(x1 * cs + x0 * sn);
}

// ---------------------------------------------------------------------------
// Flash attention, causal, head_dim_qk = 192, head_dim_v = 128.
// Grid: (S/128 q-blocks, B*H). 256 thr = 8 waves; wave owns 16 query rows.
// K tile staged via gfx1250 async global->LDS DMA (ASYNCcnt); V tile loaded
// b128 and transposed through ds_store. Per 32-key tile: 12 WMMA (QK^T) +
// 8 WMMA (P*V). Online softmax via 16-lane shuffle reductions (wave32).
// ---------------------------------------------------------------------------
__global__ __launch_bounds__(256) void mla_attn_kernel(
    const unsigned short* __restrict__ Qf, const unsigned short* __restrict__ Kf,
    const unsigned short* __restrict__ Vf, unsigned short* __restrict__ Ob,
    int S, int H)
{
  __shared__ unsigned short Kt[32 * 192];    // [key][d]
  __shared__ unsigned short Vt[128 * 32];    // [dv][key] (transposed)
  __shared__ unsigned short Pb[8 * 16 * 32]; // per-wave P relayout buffer
  const int bh = blockIdx.y, qb = blockIdx.x;
  const int tid = threadIdx.x, lane = tid & 31, w = tid >> 5;
  const int hfw = lane >> 4, lr = lane & 15;
  const unsigned short* Qh = Qf + (size_t)bh * S * 192;
  const unsigned short* Kh = Kf + (size_t)bh * S * 192;
  const unsigned short* Vh = Vf + (size_t)bh * S * 128;
  const int qrow0 = qb * 128 + w * 16;
  const unsigned kt_base = (unsigned)(uintptr_t)&Kt[0];

  AFrag qfr[6];
  {
    const unsigned short* Qrow = Qh + (size_t)(qrow0 + lr) * 192;
#pragma unroll
    for (int kk = 0; kk < 6; ++kk) {
      qfr[kk].q[0] = *(const U4*)(Qrow + kk * 32 + hfw * 8);
      qfr[kk].q[1] = *(const U4*)(Qrow + kk * 32 + 16 + hfw * 8);
    }
  }
  v8f acc[8] = {};
  float mrow[8], lrow[8];
#pragma unroll
  for (int r = 0; r < 8; ++r) { mrow[r] = -1e30f; lrow[r] = 0.f; }

  const float sc = 0.0721687836487032f;     // 1/sqrt(192)
  const int tEnd = qb * 128 + 128;
  for (int t = 0; t < tEnd; t += 32) {
    __syncthreads();                         // previous tile fully consumed
    if (t + 32 < tEnd) {                     // prefetch next K/V tiles into GL2
      __builtin_prefetch(Kh + (size_t)(t + 32) * 192 + (size_t)tid * 48, 0, 1);
      __builtin_prefetch(Vh + (size_t)(t + 32) * 128 + (size_t)tid * 16, 0, 1);
    }
#pragma unroll
    for (int i = 0; i < 3; ++i) {            // K tile: 32 keys x 192 d (async DMA)
      int idx = tid * 3 + i;
      int key = idx / 24, c = idx % 24;
      async_ld_b128(kt_base + (unsigned)(key * 192 + c * 8) * 2u,
                    Kh + (size_t)(t + key) * 192 + c * 8);
    }
#pragma unroll
    for (int i = 0; i < 2; ++i) {            // V tile -> transposed [dv][key]
      int chunk = tid * 2 + i;
      int key = chunk >> 4, dv0 = (chunk & 15) << 3;
      US8 v = *(const US8*)(Vh + (size_t)(t + key) * 128 + dv0);
#pragma unroll
      for (int j = 0; j < 8; ++j) Vt[(dv0 + j) * 32 + key] = v.s[j];
    }
    wait_async0();
    __syncthreads();

    v8f s0 = {}, s1 = {};
#pragma unroll
    for (int kk = 0; kk < 6; ++kk) {         // Q x K^T
      AFrag b0, b1;
      b0.q[0] = *(const U4*)(Kt + lr * 192 + kk * 32 + hfw * 8);
      b0.q[1] = *(const U4*)(Kt + lr * 192 + kk * 32 + 16 + hfw * 8);
      b1.q[0] = *(const U4*)(Kt + (16 + lr) * 192 + kk * 32 + hfw * 8);
      b1.q[1] = *(const U4*)(Kt + (16 + lr) * 192 + kk * 32 + 16 + hfw * 8);
      s0 = wmma_bf16(qfr[kk], b0, s0);
      s1 = wmma_bf16(qfr[kk], b1, s1);
    }

#pragma unroll
    for (int r = 0; r < 8; ++r) {            // online softmax per query row
      int row = qrow0 + hfw * 8 + r;
      float v0 = (t + lr      <= row) ? s0[r] * sc : -1e30f;
      float v1 = (t + 16 + lr <= row) ? s1[r] * sc : -1e30f;
      float mx = fmaxf(v0, v1);
#pragma unroll
      for (int off = 1; off < 16; off <<= 1) mx = fmaxf(mx, __shfl_xor(mx, off, 32));
      float mnew = fmaxf(mrow[r], mx);
      float resc = __expf(mrow[r] - mnew);
      float p0 = __expf(v0 - mnew), p1 = __expf(v1 - mnew);
      float sum = p0 + p1;
#pragma unroll
      for (int off = 1; off < 16; off <<= 1) sum += __shfl_xor(sum, off, 32);
      lrow[r] = lrow[r] * resc + sum;
      mrow[r] = mnew;
#pragma unroll
      for (int f = 0; f < 8; ++f) acc[f][r] *= resc;
      Pb[w * 512 + (hfw * 8 + r) * 32 + lr]      = f2b(p0);
      Pb[w * 512 + (hfw * 8 + r) * 32 + 16 + lr] = f2b(p1);
    }

    AFrag pf;                                // P relayout: C-frag -> A-frag
    pf.q[0] = *(const U4*)(Pb + w * 512 + lr * 32 + hfw * 8);
    pf.q[1] = *(const U4*)(Pb + w * 512 + lr * 32 + 16 + hfw * 8);
#pragma unroll
    for (int f = 0; f < 8; ++f) {            // P x V
      AFrag vb;
      vb.q[0] = *(const U4*)(Vt + (f * 16 + lr) * 32 + hfw * 8);
      vb.q[1] = *(const U4*)(Vt + (f * 16 + lr) * 32 + 16 + hfw * 8);
      acc[f] = wmma_bf16(pf, vb, acc[f]);
    }
  }

  const int bb = bh / H, hh = bh % H;        // O[b, s, h*128 + d]  bf16
#pragma unroll
  for (int f = 0; f < 8; ++f)
#pragma unroll
    for (int r = 0; r < 8; ++r) {
      int srow = qrow0 + hfw * 8 + r;
      size_t tok = (size_t)bb * S + srow;
      Ob[tok * (size_t)(H * 128) + hh * 128 + f * 16 + lr] = f2b(acc[f][r] / lrow[r]);
    }
}

// ---------------------------------------------------------------------------
// Launcher
// ---------------------------------------------------------------------------
extern "C" void kernel_launch(void* const* d_in, const int* in_sizes, int n_in,
                              void* d_out, int out_size, void* d_ws, size_t ws_size,
                              hipStream_t stream)
{
  (void)in_sizes; (void)n_in; (void)out_size; (void)ws_size;
  const float* h    = (const float*)d_in[0];
  const float* WDKV = (const float*)d_in[1];
  const float* WUK  = (const float*)d_in[2];
  const float* WKR  = (const float*)d_in[3];
  const float* WUV  = (const float*)d_in[4];
  const float* WDQ  = (const float*)d_in[5];
  const float* WUQ  = (const float*)d_in[6];
  const float* WQR  = (const float*)d_in[7];
  const float* WO   = (const float*)d_in[8];

  const int B = 2, S = 2048, H = 16, HID = 2048, LAT = 512, RD = 64;
  const int M = B * S;   // 4096 tokens

  char* ws = (char*)d_ws;
  size_t off = 0;
  float* cKV = (float*)(ws + off); off += (size_t)M * LAT * 4;       // 8 MB
  float* cQ  = (float*)(ws + off); off += (size_t)M * LAT * 4;       // 8 MB
  float* KR  = (float*)(ws + off); off += (size_t)M * RD  * 4;       // 1 MB
  float* QR  = (float*)(ws + off); off += (size_t)M * H * RD * 4;    // 16 MB
  unsigned short* Kf = (unsigned short*)(ws + off); off += (size_t)B * H * S * 192 * 2; // 24 MB
  unsigned short* Qf = (unsigned short*)(ws + off); off += (size_t)B * H * S * 192 * 2; // 24 MB
  unsigned short* Vf = (unsigned short*)(ws + off); off += (size_t)B * H * S * 128 * 2; // 16 MB
  unsigned short* Ob = (unsigned short*)(ws + off); off += (size_t)M * HID * 2;         // 16 MB

  dim3 blk(256);
  // Down projections from h (fp32 A)
  gemm_wmma_bf16<0, 0><<<dim3(LAT / 128, M / 128), blk, 0, stream>>>(
      h, WDKV, cKV, nullptr, M, LAT, HID, S, H, 0);
  gemm_wmma_bf16<0, 0><<<dim3(LAT / 128, M / 128), blk, 0, stream>>>(
      h, WDQ, cQ, nullptr, M, LAT, HID, S, H, 0);
  gemm_wmma_bf16<0, 0><<<dim3(1, M / 128), blk, 0, stream>>>(
      h, WKR, KR, nullptr, M, RD, HID, S, H, 0);
  // Up projections from latents -> head-major bf16 buffers
  gemm_wmma_bf16<0, 1><<<dim3(HID / 128, M / 128), blk, 0, stream>>>(
      cKV, WUK, nullptr, Kf, M, HID, LAT, S, H, 192);
  gemm_wmma_bf16<0, 1><<<dim3(HID / 128, M / 128), blk, 0, stream>>>(
      cKV, WUV, nullptr, Vf, M, HID, LAT, S, H, 128);
  gemm_wmma_bf16<0, 1><<<dim3(HID / 128, M / 128), blk, 0, stream>>>(
      cQ, WUQ, nullptr, Qf, M, HID, LAT, S, H, 192);
  gemm_wmma_bf16<0, 0><<<dim3((H * RD) / 128, M / 128), blk, 0, stream>>>(
      cQ, WQR, QR, nullptr, M, H * RD, LAT, S, H, 0);
  // RoPE into the 128..191 dims of K/Q head buffers
  {
    int totK = M * (RD / 2);
    rope_k_kernel<<<(totK + 255) / 256, blk, 0, stream>>>(KR, Kf, S, H, totK);
    int totQ = M * H * (RD / 2);
    rope_q_kernel<<<(totQ + 255) / 256, blk, 0, stream>>>(QR, Qf, S, H, totQ);
  }
  // Flash attention -> Ob (bf16)
  mla_attn_kernel<<<dim3(S / 128, B * H), blk, 0, stream>>>(Qf, Kf, Vf, Ob, S, H);
  // Output projection (bf16 A path) -> d_out fp32
  gemm_wmma_bf16<1, 0><<<dim3(HID / 128, M / 128), blk, 0, stream>>>(
      Ob, WO, (float*)d_out, nullptr, M, HID, HID, S, H, 0);
}
